// NLBlockND_cross_8985071583928
// MI455X (gfx1250) — matrix operand order repaired and checked by reading, not compiled, not run on hardware
//
#include <hip/hip_runtime.h>
#include <hip/hip_bf16.h>

#define NB    4
#define CCH   128
#define INTER 64
#define HW    4096

typedef __attribute__((ext_vector_type(16))) _Float16 v16h;
typedef __attribute__((ext_vector_type(8)))  float    v8f;
typedef __attribute__((ext_vector_type(4))) unsigned int u32x4;
typedef __attribute__((ext_vector_type(8))) int i32x8;
typedef __attribute__((ext_vector_type(4))) int i32x4;

#if defined(__gfx1250__) && __has_builtin(__builtin_amdgcn_tensor_load_to_lds) && \
    __has_builtin(__builtin_amdgcn_s_wait_tensorcnt)
#define USE_TDM 1
#else
#define USE_TDM 0
#endif

// ---------- fragment-layout index helpers (contiguous-k order, same for A and B) ----------
// B-frag k order used by this file (A loads match): lanes 0-15 hold K=0..15 (N=lane), lanes 16-31 hold
// K=16..31 (N=lane-16); 16 halves per lane, K ascending.
__device__ __forceinline__ size_t thetaB_idx(int n, int cK, int t) {
  int tt = t >> 4, kb = cK >> 5, kRel = cK & 31;
  int ln = (t & 15) + ((kRel >> 4) << 4);
  int hh = kRel & 15;
  return ((((size_t)n * 256 + tt) * 2 + kb) * 32 + ln) * 16 + hh;
}
__device__ __forceinline__ size_t gB_idx(int n, int t, int c) {
  int tb = t >> 5, tRel = t & 31, ct = c >> 4;
  int ln = (c & 15) + ((tRel >> 4) << 4);
  int hh = tRel & 15;
  return ((((size_t)n * 128 + tb) * 4 + ct) * 32 + ln) * 16 + hh;
}

__device__ __forceinline__ v16h lds_frag(const _Float16* base, int lane) {
  union { uint4 u[2]; v16h v; } r;
  const uint4* p = reinterpret_cast<const uint4*>(base + lane * 16);
  r.u[0] = p[0];
  r.u[1] = p[1];
  return r.v;
}

// Pure-VALU 16-lane xor-butterfly max via DPP (quad_perm xor1/xor2, then
// row_half_mirror / row_mirror which act as xor4/xor8 once quads are uniform).
template <int CTRL>
__device__ __forceinline__ float dpp_bfly_max(float v) {
  int o = __builtin_amdgcn_update_dpp(0, __builtin_bit_cast(int, v), CTRL, 0xf, 0xf, true);
  return fmaxf(v, __builtin_bit_cast(float, o));
}
__device__ __forceinline__ float rowmax16(float v) {
  v = dpp_bfly_max<0xB1>(v);   // quad_perm [1,0,3,2]  (xor 1)
  v = dpp_bfly_max<0x4E>(v);   // quad_perm [2,3,0,1]  (xor 2)
  v = dpp_bfly_max<0x141>(v);  // row_half_mirror      (xor 4 at quad granularity)
  v = dpp_bfly_max<0x140>(v);  // row_mirror           (xor 8 at octet granularity)
  return v;
}

#if USE_TDM
// 1-D TDM copy: `bytes` (multiple of 8) from global to LDS as 8-byte elements.
__device__ __forceinline__ void tdm_load_1d(unsigned lds_off, unsigned long long gaddr,
                                            unsigned bytes) {
  unsigned n8 = bytes >> 3;
  u32x4 g0;
  g0[0] = 1u;                                                   // count=1 user D#
  g0[1] = lds_off;                                              // lds_addr
  g0[2] = (unsigned)(gaddr & 0xffffffffu);                      // global_addr lo
  g0[3] = (unsigned)((gaddr >> 32) & 0x1ffffffu) | (2u << 30);  // addr hi | type=2
  i32x8 g1 = {};
  g1[0] = (3 << 16);              // data_size = 8B, no multicast/iterate/pad
  g1[1] = (int)(n8 << 16);        // tensor_dim0[15:0] at bits[63:48]
  g1[2] = (int)(1u << 16);        // tensor_dim0 hi=0 | tensor_dim1=1
  g1[3] = (int)(n8 << 16);        // tile_dim0 at bits[127:112]
  g1[4] = 1;                      // tile_dim1 = 1
  g1[5] = (int)n8;                // tensor_dim0_stride
  i32x4 z = {};
#if __clang_major__ >= 23
  __builtin_amdgcn_tensor_load_to_lds(g0, g1, z, z, (i32x8){}, 0);
#else
  __builtin_amdgcn_tensor_load_to_lds(g0, g1, z, z, 0);
#endif
}
#endif

// ---------- kernel 1: theta (B-frag layout) + phi (row-major A layout) ----------
__global__ __launch_bounds__(256)
void proj_kernel(const float* __restrict__ xThis, const float* __restrict__ xOther,
                 const float* __restrict__ thetaW, const float* __restrict__ thetaBias,
                 const float* __restrict__ phiW, const float* __restrict__ phiBias,
                 _Float16* __restrict__ phiA, _Float16* __restrict__ thetaB) {
  __shared__ float xtS[CCH * 64];
  __shared__ float xoS[CCH * 64];
  const int tid = threadIdx.x;
  const int n  = blockIdx.x >> 6;
  const int t0 = (blockIdx.x & 63) * 64;
  for (int i = tid; i < CCH * 64; i += 256) {
    int cc = i >> 6, tl = i & 63;
    size_t g = ((size_t)(n * CCH + cc)) * HW + t0 + tl;
    xtS[i] = xThis[g];
    xoS[i] = xOther[g];
  }
  __syncthreads();
  for (int i = 0; i < 16; ++i) {
    int idx = tid + i * 256;
    int tl = idx & 63;           // varies per lane -> conflict-free LDS, coalesced stores
    int co = idx >> 6;           // uniform per wave -> scalar weight loads
    float at = thetaBias[co];
    float ap = phiBias[co];
#pragma unroll 4
    for (int cc = 0; cc < CCH; ++cc) {
      at += thetaW[co * CCH + cc] * xtS[cc * 64 + tl];
      ap += phiW[co * CCH + cc]   * xoS[cc * 64 + tl];
    }
    int t = t0 + tl;
    phiA[((size_t)n * HW + t) * INTER + co] = (_Float16)ap;
    thetaB[thetaB_idx(n, co, t)] = (_Float16)at;
  }
}

// ---------- kernel 2: hidden = SiLU(x_other @ lin_w^T + b) -> g (B-frag) + gate ----------
__global__ __launch_bounds__(256)
void hidden_kernel(const float* __restrict__ xOther,
                   const float* __restrict__ linW, const float* __restrict__ linB,
                   _Float16* __restrict__ gB, float* __restrict__ gateT) {
  __shared__ float xoS[64 * 64];
  __shared__ float lwS[64 * 65];   // padded to kill bank conflicts
  const int tid = threadIdx.x;
  const int n = blockIdx.x >> 7;
  const int c = blockIdx.x & 127;
  for (int i = tid; i < 4096; i += 256)
    xoS[i] = xOther[((size_t)(n * CCH + c)) * HW + i];
  for (int i = tid; i < 4096; i += 256) {
    int o = i >> 6, w = i & 63;
    lwS[o * 65 + w] = linW[i];
  }
  __syncthreads();
  for (int i = 0; i < 16; ++i) {
    int idx = tid + i * 256;
    int o = idx & 63;            // varies per lane
    int h = idx >> 6;            // uniform per wave -> xoS broadcast
    float a = linB[o];
#pragma unroll 4
    for (int w = 0; w < 64; ++w)
      a += xoS[h * 64 + w] * lwS[o * 65 + w];
    a = a / (1.0f + __expf(-a)); // SiLU
    int t = h * 64 + o;
    if (c < INTER) {
      gB[gB_idx(n, t, c)] = (_Float16)a;
    } else {
      gateT[((size_t)n * HW + t) * INTER + (c - INTER)] = a;
    }
  }
}

// ---------- kernel 3: fused flash attention (QK^T -> online softmax -> PV) ----------
__global__ __launch_bounds__(256)
void flash_attn_kernel(const _Float16* __restrict__ phiA,
                       const _Float16* __restrict__ thetaB,
                       const _Float16* __restrict__ gB,
                       const float* __restrict__ gateT,
                       float* __restrict__ yOut) {
  __shared__ __align__(16) _Float16 stage[2][4096];  // [theta frag x4 | g frag x4]
  __shared__ __align__(16) _Float16 pScr[8][512];    // per-wave P transpose scratch

  const int tid  = threadIdx.x;
  const int lane = tid & 31;
  const int wave = tid >> 5;
  const int n      = blockIdx.x >> 5;
  const int sGroup = blockIdx.x & 31;
  const int s0 = (sGroup * 8 + wave) * 16;

  // A fragments: phi rows s0..s0+15, K = channel 0..63 (two 16x32 frags)
  const int aRow = s0 + (lane & 15);
  const int aCb  = (lane >> 4) * 16;
  const _Float16* ap = phiA + ((size_t)n * HW + aRow) * INTER;
  v16h aFrag[2];
  {
    union { uint4 u[2]; v16h v; } r0, r1;
    r0.u[0] = *reinterpret_cast<const uint4*>(ap + aCb);
    r0.u[1] = *reinterpret_cast<const uint4*>(ap + aCb + 8);
    r1.u[0] = *reinterpret_cast<const uint4*>(ap + 32 + aCb);
    r1.u[1] = *reinterpret_cast<const uint4*>(ap + 32 + aCb + 8);
    aFrag[0] = r0.v; aFrag[1] = r1.v;
  }

  // All-ones B fragment: rowsum(P) = P @ ones computed on the WMMA pipe.
  v16h ones;
#pragma unroll
  for (int i = 0; i < 16; ++i) ones[i] = (_Float16)1.0f;

  float m[8];
  v8f acc[4] = {};
  v8f lAcc = {};
#pragma unroll
  for (int r = 0; r < 8; ++r) m[r] = -1e30f;

  const uint4* thSrc = reinterpret_cast<const uint4*>(thetaB + (size_t)n * 256 * 1024);
  const uint4* gSrc  = reinterpret_cast<const uint4*>(gB     + (size_t)n * 128 * 2048);

#if USE_TDM
  auto stageCopy = [&](int buf, int tb) {
    // Two 4KB 1-D TDM descriptor copies, issued once per workgroup (wave 0).
    unsigned ldsT = (unsigned)(unsigned long long)&stage[buf][0];
    unsigned ldsG = (unsigned)(unsigned long long)&stage[buf][2048];
    tdm_load_1d(ldsT, (unsigned long long)(thSrc + (size_t)tb * 256), 4096);
    tdm_load_1d(ldsG, (unsigned long long)(gSrc  + (size_t)tb * 256), 4096);
  };
#else
  auto stageCopy = [&](int buf, int tb) {
    uint4 tv = thSrc[(size_t)tb * 256 + tid];
    uint4 gv = gSrc [(size_t)tb * 256 + tid];
    reinterpret_cast<uint4*>(&stage[buf][0])[tid]    = tv;
    reinterpret_cast<uint4*>(&stage[buf][2048])[tid] = gv;
  };
#endif

#if USE_TDM
  if (wave == 0) stageCopy(0, 0);
#else
  stageCopy(0, 0);
#endif

  for (int tb = 0; tb < 128; ++tb) {
    const int buf = tb & 1;
#if USE_TDM
    if (wave == 0) __builtin_amdgcn_s_wait_tensorcnt(0);  // stage[buf] landed
    __syncthreads();
    if (tb + 1 < 128 && wave == 0) stageCopy(buf ^ 1, tb + 1);
#else
    __syncthreads();                       // stage[buf] ready; stage[buf^1] free
    if (tb + 1 < 128) stageCopy(buf ^ 1, tb + 1);
    if (tb + 2 < 128) {                    // gfx1250 global_prefetch of step+2
      __builtin_prefetch(thSrc + (size_t)(tb + 2) * 256 + tid, 0, 1);
      __builtin_prefetch(gSrc  + (size_t)(tb + 2) * 256 + tid, 0, 1);
    }
#endif

    const _Float16* th = &stage[buf][0];
    const _Float16* gg = &stage[buf][2048];

    // ---- QK^T: two 16x16 score tiles over K=64 channels ----
    v16h b00 = lds_frag(th + 0 * 512, lane);
    v16h b01 = lds_frag(th + 1 * 512, lane);
    v16h b10 = lds_frag(th + 2 * 512, lane);
    v16h b11 = lds_frag(th + 3 * 512, lane);
    v8f st0 = {}, st1 = {};
    st0 = __builtin_amdgcn_wmma_f32_16x16x32_f16(false, aFrag[0], false, b00, (short)0, st0, false, false);
    st0 = __builtin_amdgcn_wmma_f32_16x16x32_f16(false, aFrag[1], false, b01, (short)0, st0, false, false);
    st1 = __builtin_amdgcn_wmma_f32_16x16x32_f16(false, aFrag[0], false, b10, (short)0, st1, false, false);
    st1 = __builtin_amdgcn_wmma_f32_16x16x32_f16(false, aFrag[1], false, b11, (short)0, st1, false, false);

    // ---- online softmax over 32 columns: max via DPP butterfly (pure VALU) ----
    float p0[8], p1[8];
#pragma unroll
    for (int r = 0; r < 8; ++r) {
      float v = rowmax16(fmaxf(st0[r], st1[r]));
      float mn = fmaxf(m[r], v);
      float sc = __expf(m[r] - mn);
      m[r] = mn;
      p0[r] = __expf(st0[r] - mn);
      p1[r] = __expf(st1[r] - mn);
      lAcc[r] *= sc;
      acc[0][r] *= sc; acc[1][r] *= sc; acc[2][r] *= sc; acc[3][r] *= sc;
    }

    // ---- transpose P (C-layout) -> A-fragment layout via per-wave LDS ----
    _Float16* ps = pScr[wave];
    const int rowB = (lane >> 4) * 8;
    const int colL = lane & 15;
#pragma unroll
    for (int r = 0; r < 8; ++r) {
      ps[(rowB + r) * 32 + colL]      = (_Float16)p0[r];
      ps[(rowB + r) * 32 + 16 + colL] = (_Float16)p1[r];
    }
    v16h pa;
    {
      union { uint4 u[2]; v16h v; } r;
      const _Float16* pr = ps + (lane & 15) * 32 + (lane >> 4) * 16;
      r.u[0] = *reinterpret_cast<const uint4*>(pr);
      r.u[1] = *reinterpret_cast<const uint4*>(pr + 8);
      pa = r.v;
    }

    // ---- denominator: lAcc += rowsum(P) on the matrix pipe ----
    lAcc = __builtin_amdgcn_wmma_f32_16x16x32_f16(false, pa, false, ones, (short)0, lAcc, false, false);

    // ---- PV: Y[16x64] += P[16x32] @ G[32x64] ----
#pragma unroll
    for (int ct = 0; ct < 4; ++ct) {
      v16h bg = lds_frag(gg + ct * 512, lane);
      acc[ct] = __builtin_amdgcn_wmma_f32_16x16x32_f16(false, pa, false, bg, (short)0, acc[ct], false, false);
    }
  }

  // ---- epilogue: normalize, gate, store y[n][s][c] ----
  float rinv[8];
#pragma unroll
  for (int r = 0; r < 8; ++r) rinv[r] = 1.0f / lAcc[r];
  const int rowOff = (lane >> 4) * 8;
#pragma unroll
  for (int ct = 0; ct < 4; ++ct) {
#pragma unroll
    for (int r = 0; r < 8; ++r) {
      int s = s0 + rowOff + r;
      int c = ct * 16 + (lane & 15);
      size_t gi = ((size_t)n * HW + s) * INTER + c;
      yOut[gi] = acc[ct][r] * rinv[r] * gateT[gi];
    }
  }
}

// ---------- kernel 4: Wz projection + BatchNorm + residual ----------
__global__ __launch_bounds__(256)
void wz_out_kernel(const float* __restrict__ yArr, const float* __restrict__ wzW,
                   const float* __restrict__ wzB, const float* __restrict__ gamma,
                   const float* __restrict__ beta, const float* __restrict__ mean,
                   const float* __restrict__ var, const float* __restrict__ xThis,
                   float* __restrict__ out) {
  __shared__ float wzS[CCH * INTER];
  __shared__ float yS[32 * 65];     // padded
  const int tid = threadIdx.x;
  const int n  = blockIdx.x >> 7;
  const int s0 = (blockIdx.x & 127) * 32;
  for (int i = tid; i < CCH * INTER; i += 256) wzS[i] = wzW[i];
  for (int i = tid; i < 32 * INTER; i += 256) {
    int si = i >> 6, cc = i & 63;
    yS[si * 65 + cc] = yArr[((size_t)n * HW + s0 + si) * INTER + cc];
  }
  __syncthreads();
  const int si = tid & 31;          // varies per lane -> coalesced global access
  for (int i = 0; i < 16; ++i) {
    int co = (tid >> 5) + i * 8;    // uniform per wave -> wzS broadcast
    float a = wzB[co];
#pragma unroll 4
    for (int cc = 0; cc < INTER; ++cc)
      a += wzS[co * INTER + cc] * yS[si * 65 + cc];
    float sc = gamma[co] * rsqrtf(var[co] + 1e-5f);
    float z = (a - mean[co]) * sc + beta[co];
    size_t gi = ((size_t)(n * CCH + co)) * HW + s0 + si;
    out[gi] = z + xThis[gi];
  }
}

extern "C" void kernel_launch(void* const* d_in, const int* in_sizes, int n_in,
                              void* d_out, int out_size, void* d_ws, size_t ws_size,
                              hipStream_t stream) {
  (void)in_sizes; (void)n_in; (void)out_size; (void)ws_size;
  const float* xThis     = (const float*)d_in[0];
  const float* xOther    = (const float*)d_in[1];
  const float* linW      = (const float*)d_in[2];
  const float* linB      = (const float*)d_in[3];
  const float* thetaW    = (const float*)d_in[4];
  const float* thetaBias = (const float*)d_in[5];
  const float* phiW      = (const float*)d_in[6];
  const float* phiBias   = (const float*)d_in[7];
  const float* wzW       = (const float*)d_in[8];
  const float* wzB       = (const float*)d_in[9];
  const float* gamma     = (const float*)d_in[10];
  const float* beta      = (const float*)d_in[11];
  const float* mean      = (const float*)d_in[12];
  const float* var       = (const float*)d_in[13];
  float* out = (float*)d_out;

  char* ws = (char*)d_ws;
  _Float16* phiA   = (_Float16*)(ws);                          // 2 MiB  f16 [n][s][c]
  _Float16* thetaB = (_Float16*)(ws + ((size_t)2 << 20));      // 2 MiB  f16 B-frag
  _Float16* gB     = (_Float16*)(ws + ((size_t)4 << 20));      // 2 MiB  f16 B-frag
  float*    gateT  = (float*)  (ws + ((size_t)6 << 20));       // 4 MiB  f32 [n][s][c]
  float*    yArr   = (float*)  (ws + ((size_t)10 << 20));      // 4 MiB  f32 [n][s][c]

  proj_kernel<<<dim3(NB * 64), dim3(256), 0, stream>>>(
      xThis, xOther, thetaW, thetaBias, phiW, phiBias, phiA, thetaB);
  hidden_kernel<<<dim3(NB * CCH), dim3(256), 0, stream>>>(
      xOther, linW, linB, gB, gateT);
  flash_attn_kernel<<<dim3(NB * 32), dim3(256), 0, stream>>>(
      phiA, thetaB, gB, gateT, yArr);
  wz_out_kernel<<<dim3(NB * 128), dim3(256), 0, stream>>>(
      yArr, wzW, wzB, gamma, beta, mean, var, xThis, out);
}
